// GRUD_CLEAN_22763326669177
// MI455X (gfx1250) — compile-verified, matrix operand order checked by reading, not run on hardware
//
#include <hip/hip_runtime.h>
#include <hip/hip_bf16.h>

// ---------------------------------------------------------------------------
// GRU-D forward for gfx1250 (MI455X): bf16 WMMA scan, batch-partitioned.
// B=256, T=256, D=128, H=1024, O=64.
// 16 batch groups (16 rows each) x 8 WGs/group (128 H-cols each) = 128 WGs.
// Each WG: 256 threads = 8 wave32; each wave owns one 16x16 H-tile in WMMA
// C-layout registers for the whole scan. Cross-WG h exchange goes through L2
// and is staged into LDS with the CDNA5 async-copy path when available.
// ---------------------------------------------------------------------------

typedef __attribute__((ext_vector_type(16))) __bf16 v16bf;
typedef __attribute__((ext_vector_type(8)))  __bf16 v8bf;
typedef __attribute__((ext_vector_type(8)))  float  v8f;
typedef __attribute__((ext_vector_type(4)))  int    v4i;

#define GB  256   // batch
#define GT  256   // time
#define GD  128   // input dim
#define GH  1024  // hidden
#define GO  64    // out
#define NGROUP 16
#define WPG    8   // workgroups per group
#define SLICE  128 // H columns per WG
#define NWAVE  8
#define NTHR   256

#if defined(__has_builtin)
#if __has_builtin(__builtin_amdgcn_global_load_async_to_lds_b128)
#define USE_ASYNC_LDS 1
#endif
#if __has_builtin(__builtin_amdgcn_s_wait_asynccnt)
#define HAVE_WAIT_ASYNC_BUILTIN 1
#endif
#endif

__device__ __forceinline__ void wait_asynccnt0() {
#if defined(HAVE_WAIT_ASYNC_BUILTIN)
  __builtin_amdgcn_s_wait_asynccnt(0);
#elif defined(USE_ASYNC_LDS)
  asm volatile("s_wait_asynccnt 0x0" ::: "memory");
#endif
}

__device__ __forceinline__ v8f zero8() {
  v8f z;
#pragma unroll
  for (int i = 0; i < 8; ++i) z[i] = 0.f;
  return z;
}

__device__ __forceinline__ v8f wmma_bf16(v16bf a, v16bf b, v8f c) {
  return __builtin_amdgcn_wmma_f32_16x16x32_bf16(
      /*neg_a=*/false, a, /*neg_b=*/false, b,
      /*c_mod=*/(short)0, c, /*reuse_a=*/false, /*reuse_b=*/false);
}

// Load a 16x32 fragment (A role: rows = lane%16 over M; B role: rows = lane%16
// over N of an N-major matrix). Per lane: two contiguous 16B chunks.
//   j=0..7  -> K = k0 + 8*(lane/16) + j
//   j=8..15 -> K = k0 + 16 + 8*(lane/16) + (j-8)
__device__ __forceinline__ v16bf ldfrag(const __bf16* base, int row0, int ldk, int k0) {
  const int lane = threadIdx.x & 31;
  const int r = row0 + (lane & 15);
  const int h = lane >> 4;
  const __bf16* p = base + (size_t)r * ldk + k0 + 8 * h;
  v8bf lo = *(const v8bf*)(p);
  v8bf hi = *(const v8bf*)(p + 16);
  v16bf out;
#pragma unroll
  for (int i = 0; i < 8; ++i) { out[i] = lo[i]; out[i + 8] = hi[i]; }
  return out;
}

// Stage a [16 x GH] bf16 tile (32 KB) from global (L2) into LDS.
__device__ __forceinline__ void stage_to_lds(const __bf16* gsrc, __bf16* ldst, int tid) {
#if defined(USE_ASYNC_LDS)
  typedef __attribute__((address_space(1))) v4i gv4i;
  typedef __attribute__((address_space(3))) v4i sv4i;
#pragma unroll
  for (int i = tid; i < (16 * GH) / 8; i += NTHR) {  // 16-byte chunks per lane
    __builtin_amdgcn_global_load_async_to_lds_b128(
        (gv4i*)(gsrc + (size_t)i * 8), (sv4i*)(ldst + (size_t)i * 8), 0, 0);
  }
  wait_asynccnt0();
#else
  const uint4* src = (const uint4*)gsrc;
  uint4* dst = (uint4*)ldst;
  for (int i = tid; i < (16 * GH * 2) / 16; i += NTHR) dst[i] = src[i];
#endif
}

// Sense-reversing barrier among WPG workgroups of one batch group.
__device__ __forceinline__ void group_barrier(unsigned* bar) {
  __syncthreads();
  if (threadIdx.x == 0) {
    __threadfence();
    unsigned gen = __hip_atomic_load(&bar[1], __ATOMIC_ACQUIRE, __HIP_MEMORY_SCOPE_AGENT);
    unsigned arr = __hip_atomic_fetch_add(&bar[0], 1u, __ATOMIC_ACQ_REL, __HIP_MEMORY_SCOPE_AGENT);
    if (arr == WPG - 1) {
      __hip_atomic_store(&bar[0], 0u, __ATOMIC_RELAXED, __HIP_MEMORY_SCOPE_AGENT);
      __hip_atomic_fetch_add(&bar[1], 1u, __ATOMIC_RELEASE, __HIP_MEMORY_SCOPE_AGENT);
    } else {
      while (__hip_atomic_load(&bar[1], __ATOMIC_ACQUIRE, __HIP_MEMORY_SCOPE_AGENT) == gen)
        __builtin_amdgcn_s_sleep(1);
    }
    __threadfence();
  }
  __syncthreads();
}

// --------------------------- prepass kernels -------------------------------

// dst (bf16, N-major [N][K]) = transpose of src (f32, row-major [K][N])
__global__ void k_transpose_bf16(unsigned short* dst_, const float* src, int K, int N) {
  __bf16* dst = (__bf16*)dst_;
  int i = blockIdx.x * NTHR + threadIdx.x;
  if (i >= K * N) return;
  int n = i / K, k = i % K;
  dst[i] = (__bf16)src[(size_t)k * N + n];
}

// Bcat (bf16, N-major [1024][256]): rows k<128 from Wg^T, k>=128 from Vg^T.
__global__ void k_bcat(unsigned short* dst_, const float* Wg, const float* Vg) {
  __bf16* dst = (__bf16*)dst_;
  int i = blockIdx.x * NTHR + threadIdx.x;
  if (i >= GH * 256) return;
  int n = i >> 8, k = i & 255;
  float v = (k < GD) ? Wg[(size_t)k * GH + n] : Vg[(size_t)(k - GD) * GH + n];
  dst[i] = (__bf16)v;
}

__global__ void k_init_out(float* out, const float* dec_b, unsigned* bars) {
  int i = blockIdx.x * NTHR + threadIdx.x;
  if (i < GB * GO) out[i] = dec_b[i & (GO - 1)];
  if (i < NGROUP * 64) bars[i] = 0u;
}

// ----------------------------- scan kernel ---------------------------------

__global__ __launch_bounds__(NTHR) void k_grud_scan(
    const float* __restrict__ x, const float* __restrict__ delta, const float* __restrict__ m,
    const float* __restrict__ Wgx, const float* __restrict__ bgx, const float* __restrict__ bgh,
    const float* __restrict__ bz, const float* __restrict__ br, const float* __restrict__ bh,
    const unsigned short* UtZ_, const unsigned short* UtR_, const unsigned short* UtH_,
    const unsigned short* BcZ_, const unsigned short* BcR_, const unsigned short* BcH_,
    const unsigned short* WghT_, const unsigned short* decWT_,
    unsigned short* a1buf_, unsigned short* rhbuf_, unsigned* bars, float* out) {
  const __bf16* UtZ = (const __bf16*)UtZ_;
  const __bf16* UtR = (const __bf16*)UtR_;
  const __bf16* UtH = (const __bf16*)UtH_;
  const __bf16* BcZ = (const __bf16*)BcZ_;
  const __bf16* BcR = (const __bf16*)BcR_;
  const __bf16* BcH = (const __bf16*)BcH_;
  const __bf16* WghT = (const __bf16*)WghT_;
  const __bf16* decWT = (const __bf16*)decWT_;

  const int group = blockIdx.x / WPG;
  const int wg    = blockIdx.x % WPG;
  const int tid   = threadIdx.x;
  const int lane  = tid & 31;
  const int wave  = tid >> 5;
  const int lcol  = lane & 15;
  const int lhalf = lane >> 4;
  const int gcol0 = wg * SLICE + wave * 16;  // this wave's H-tile base column
  const int col   = gcol0 + lcol;

  __bf16* A1 = (__bf16*)a1buf_ + (size_t)group * 16 * GH;
  __bf16* RH = (__bf16*)rhbuf_ + (size_t)group * 16 * GH;
  unsigned* bar = bars + group * 64;

  __shared__ __bf16 s_xcat[16 * 256];   // [x' | 1-m] A-matrix, 8 KB
  __shared__ __bf16 s_mA[16 * GD];      // m A-matrix, 4 KB
  __shared__ __bf16 s_stage[16 * GH];   // staged A1 / RH, 32 KB

  // per-lane biases for this wave's fixed columns
  const float vbz = bz[col], vbr = br[col], vbh = bh[col], vbgh = bgh[col];

  v8f hreg = zero8();  // h state, WMMA C layout: row = v + 8*lhalf, col = col

#pragma unroll 1
  for (int t = 0; t < GT; ++t) {
    __syncthreads();  // protect s_xcat/s_mA rebuild
    // ---- build x' / (1-m) / m A-matrices for this timestep ----
    for (int i = tid; i < 16 * GD; i += NTHR) {
      int rrow = i >> 7, d = i & (GD - 1);
      size_t gidx = (((size_t)(group * 16 + rrow)) * GT + t) * GD + d;
      float xv = x[gidx], dv = delta[gidx], mv = m[gidx];
      float decay = dv * Wgx[d] + bgx[d];
      float xp = (mv > 0.f) ? (1.f - decay) * 0.001f : xv;
      s_xcat[rrow * 256 + d] = (__bf16)xp;
      s_xcat[rrow * 256 + GD + d] = (__bf16)(1.f - mv);
      s_mA[rrow * GD + d] = (__bf16)mv;
    }
    __syncthreads();

    // ---- gamma_h for own tile, then h_dec = gamma_h * h ----
    v8f acc = zero8();
#pragma unroll
    for (int k = 0; k < GD; k += 32) {
      v16bf a = ldfrag(s_mA, 0, GD, k);
      v16bf b = ldfrag(WghT, gcol0, GD, k);
      acc = wmma_bf16(a, b, acc);
    }
    v8f hdec;
#pragma unroll
    for (int v = 0; v < 8; ++v) {
      float g = __expf(-fmaxf(0.f, acc[v] + vbgh));
      hdec[v] = g * hreg[v];
    }
#pragma unroll
    for (int v = 0; v < 8; ++v)
      A1[(size_t)(v + 8 * lhalf) * GH + col] = (__bf16)hdec[v];
    group_barrier(bar);

    // ---- stage A1 (16x1024 bf16 = 32 KB) into LDS ----
    stage_to_lds(A1, s_stage, tid);
    __syncthreads();

    // ---- phase 1: z and r gates ----
    v8f accZ = zero8(), accR = zero8();
#pragma unroll
    for (int k = 0; k < 256; k += 32) {  // input-side (x'|1-m) @ [W;V]
      v16bf a = ldfrag(s_xcat, 0, 256, k);
      accZ = wmma_bf16(a, ldfrag(BcZ, gcol0, 256, k), accZ);
      accR = wmma_bf16(a, ldfrag(BcR, gcol0, 256, k), accR);
    }
#pragma unroll 4
    for (int k = 0; k < GH; k += 32) {   // h_dec @ U_z / U_r
      v16bf a = ldfrag(s_stage, 0, GH, k);
      if (k + 128 < GH) {
        __builtin_prefetch(UtZ + (size_t)(gcol0 + lcol) * GH + k + 128, 0, 1);
        __builtin_prefetch(UtR + (size_t)(gcol0 + lcol) * GH + k + 128, 0, 1);
      }
      accZ = wmma_bf16(a, ldfrag(UtZ, gcol0, GH, k), accZ);
      accR = wmma_bf16(a, ldfrag(UtR, gcol0, GH, k), accR);
    }
    v8f zreg;
#pragma unroll
    for (int v = 0; v < 8; ++v) {
      float zz = 1.f / (1.f + __expf(-(accZ[v] + vbz)));
      float rr = 1.f / (1.f + __expf(-(accR[v] + vbr)));
      zreg[v] = zz;
      RH[(size_t)(v + 8 * lhalf) * GH + col] = (__bf16)(rr * hdec[v]);
    }
    group_barrier(bar);

    // ---- stage RH into LDS ----
    stage_to_lds(RH, s_stage, tid);
    __syncthreads();

    // ---- phase 2: candidate h_tilde, combine ----
    v8f accH = zero8();
#pragma unroll
    for (int k = 0; k < 256; k += 32) {
      v16bf a = ldfrag(s_xcat, 0, 256, k);
      accH = wmma_bf16(a, ldfrag(BcH, gcol0, 256, k), accH);
    }
#pragma unroll 4
    for (int k = 0; k < GH; k += 32) {
      v16bf a = ldfrag(s_stage, 0, GH, k);
      if (k + 128 < GH)
        __builtin_prefetch(UtH + (size_t)(gcol0 + lcol) * GH + k + 128, 0, 1);
      accH = wmma_bf16(a, ldfrag(UtH, gcol0, GH, k), accH);
    }
#pragma unroll
    for (int v = 0; v < 8; ++v) {
      float ht = tanhf(accH[v] + vbh);
      hreg[v] = (1.f - zreg[v]) * hdec[v] + zreg[v] * ht;
    }
  }

  // ---- decoder: out += h_T(this WG's K-slice) @ dec_W ----
  __syncthreads();
#pragma unroll
  for (int v = 0; v < 8; ++v)
    s_mA[(v + 8 * lhalf) * GD + wave * 16 + lcol] = (__bf16)hreg[v];
  __syncthreads();
  if (wave < 4) {
    const int o0 = wave * 16;
    v8f acc = zero8();
#pragma unroll
    for (int k = 0; k < SLICE; k += 32) {
      v16bf a = ldfrag(s_mA, 0, GD, k);
      v16bf b = ldfrag(decWT, o0, GH, wg * SLICE + k);
      acc = wmma_bf16(a, b, acc);
    }
#pragma unroll
    for (int v = 0; v < 8; ++v) {
      float val = acc[v];
      atomicAdd(&out[(size_t)(group * 16 + v + 8 * lhalf) * GO + o0 + lcol], val);
    }
  }
}

// ------------------------------- host side ---------------------------------

extern "C" void kernel_launch(void* const* d_in, const int* in_sizes, int n_in,
                              void* d_out, int out_size, void* d_ws, size_t ws_size,
                              hipStream_t stream) {
  (void)in_sizes; (void)n_in; (void)out_size; (void)ws_size;
  const float* x     = (const float*)d_in[0];
  const float* delta = (const float*)d_in[1];
  const float* m     = (const float*)d_in[2];
  const float* W_r   = (const float*)d_in[3];
  const float* U_r   = (const float*)d_in[4];
  const float* V_r   = (const float*)d_in[5];
  const float* b_r   = (const float*)d_in[6];
  const float* W_z   = (const float*)d_in[7];
  const float* U_z   = (const float*)d_in[8];
  const float* V_z   = (const float*)d_in[9];
  const float* b_z   = (const float*)d_in[10];
  const float* W     = (const float*)d_in[11];
  const float* U     = (const float*)d_in[12];
  const float* V     = (const float*)d_in[13];
  const float* b     = (const float*)d_in[14];
  const float* Wgx   = (const float*)d_in[15];
  const float* bgx   = (const float*)d_in[16];
  const float* Wgh   = (const float*)d_in[17];
  const float* bgh   = (const float*)d_in[18];
  const float* decW  = (const float*)d_in[19];
  const float* decb  = (const float*)d_in[20];
  float* out = (float*)d_out;

  char* ws = (char*)d_ws;
  size_t off = 0;
  auto take = [&](size_t bytes) { size_t r = off; off += (bytes + 255) & ~(size_t)255; return r; };
  unsigned short* UtZ   = (unsigned short*)(ws + take((size_t)GH * GH * 2));
  unsigned short* UtR   = (unsigned short*)(ws + take((size_t)GH * GH * 2));
  unsigned short* UtH   = (unsigned short*)(ws + take((size_t)GH * GH * 2));
  unsigned short* BcZ   = (unsigned short*)(ws + take((size_t)GH * 256 * 2));
  unsigned short* BcR   = (unsigned short*)(ws + take((size_t)GH * 256 * 2));
  unsigned short* BcH   = (unsigned short*)(ws + take((size_t)GH * 256 * 2));
  unsigned short* WghT  = (unsigned short*)(ws + take((size_t)GH * GD * 2));
  unsigned short* decWT = (unsigned short*)(ws + take((size_t)GO * GH * 2));
  unsigned short* A1    = (unsigned short*)(ws + take((size_t)NGROUP * 16 * GH * 2));
  unsigned short* RH    = (unsigned short*)(ws + take((size_t)NGROUP * 16 * GH * 2));
  unsigned*       bars  = (unsigned*)(ws + take((size_t)NGROUP * 64 * 4));

  const int g1M = (GH * GH + NTHR - 1) / NTHR;
  k_transpose_bf16<<<g1M, NTHR, 0, stream>>>(UtZ, U_z, GH, GH);
  k_transpose_bf16<<<g1M, NTHR, 0, stream>>>(UtR, U_r, GH, GH);
  k_transpose_bf16<<<g1M, NTHR, 0, stream>>>(UtH, U, GH, GH);
  k_transpose_bf16<<<(GD * GH + NTHR - 1) / NTHR, NTHR, 0, stream>>>(WghT, Wgh, GD, GH);
  k_transpose_bf16<<<(GH * GO + NTHR - 1) / NTHR, NTHR, 0, stream>>>(decWT, decW, GH, GO);
  const int gcat = (GH * 256 + NTHR - 1) / NTHR;
  k_bcat<<<gcat, NTHR, 0, stream>>>(BcZ, W_z, V_z);
  k_bcat<<<gcat, NTHR, 0, stream>>>(BcR, W_r, V_r);
  k_bcat<<<gcat, NTHR, 0, stream>>>(BcH, W, V);
  k_init_out<<<(GB * GO + NTHR - 1) / NTHR, NTHR, 0, stream>>>(out, decb, bars);

  k_grud_scan<<<NGROUP * WPG, NTHR, 0, stream>>>(
      x, delta, m, Wgx, bgx, bgh, b_z, b_r, b,
      UtZ, UtR, UtH, BcZ, BcR, BcH, WghT, decWT, A1, RH, bars, out);
}
